// TransformerEncoderLayer_Dist_12163347382627
// MI455X (gfx1250) — matrix operand order repaired and checked by reading, not compiled, not run on hardware
//
#include <hip/hip_runtime.h>

#define BB  2
#define EE  256
#define HH  8
#define TT  16
#define SS  512
#define DHH 32
#define FF  2048
#define TSZ (TT*SS)   // 8192 positions per batch

typedef __attribute__((ext_vector_type(8)))  float  v8f;
typedef __attribute__((ext_vector_type(16))) __bf16 v16bf;

union Frag { unsigned int u[8]; v16bf v; };

// ---- async global->LDS path (gfx1250), with compile-safe fallback ----------
#if defined(__has_builtin)
#if __has_builtin(__builtin_amdgcn_global_load_async_to_lds_b128) && \
    __has_builtin(__builtin_amdgcn_s_wait_asynccnt)
#define USE_ASYNC 1
#endif
#endif
#ifndef USE_ASYNC
#define USE_ASYNC 0
#endif

#if USE_ASYNC
typedef __attribute__((vector_size(16))) int v4i_;
typedef __attribute__((address_space(1))) v4i_ GV4;   // global int4
typedef __attribute__((address_space(3))) v4i_ LV4;   // LDS int4
__device__ __forceinline__ void async_cp16(const void* g, void* l) {
    __builtin_amdgcn_global_load_async_to_lds_b128((GV4*)g, (LV4*)l, 0, 0);
}
__device__ __forceinline__ void async_wait0() {
    __builtin_amdgcn_s_wait_asynccnt(0);
}
#else
__device__ __forceinline__ void async_wait0() {}
#endif

__device__ __forceinline__ unsigned short f2bf(float f) {
    unsigned int x = __float_as_uint(f);
    x += 0x7FFFu + ((x >> 16) & 1u);          // round-to-nearest-even
    return (unsigned short)(x >> 16);
}

// ---------------------------------------------------------------------------
// Weight fp32 -> bf16 conversion
// ---------------------------------------------------------------------------
__global__ void cvt_bf16_kernel(const float* __restrict__ in,
                                unsigned short* __restrict__ out, int n) {
    int i = blockIdx.x * 256 + threadIdx.x;
    if (i < n) out[i] = f2bf(in[i]);
}

// ---------------------------------------------------------------------------
// LayerNorm over E at each (b,t,s); outputs bf16 src2 and qk = src2 + pos
// ---------------------------------------------------------------------------
__global__ void ln1_kernel(const float* __restrict__ enc, const float* __restrict__ pos,
                           const float* __restrict__ g,   const float* __restrict__ bt,
                           unsigned short* __restrict__ qk, unsigned short* __restrict__ s2) {
    int p = blockIdx.x * blockDim.x + threadIdx.x;
    if (p >= BB * TSZ) return;
    int b = p / TSZ, r = p % TSZ;
    size_t base = (size_t)b * EE * TSZ + r;
    float sum = 0.f, sq = 0.f;
    for (int e = 0; e < EE; ++e) {
        float x = enc[base + (size_t)e * TSZ];
        sum += x; sq += x * x;
    }
    float mu = sum * (1.f / EE);
    float var = sq * (1.f / EE) - mu * mu;
    float rs = rsqrtf(var + 1e-5f);
    for (int e = 0; e < EE; ++e) {
        size_t idx = base + (size_t)e * TSZ;
        float y = (enc[idx] - mu) * rs * g[e] + bt[e];
        s2[idx] = f2bf(y);
        qk[idx] = f2bf(y + pos[idx]);
    }
}

__global__ void ln2_kernel(const float* __restrict__ xin,
                           const float* __restrict__ g, const float* __restrict__ bt,
                           unsigned short* __restrict__ s2) {
    int p = blockIdx.x * blockDim.x + threadIdx.x;
    if (p >= BB * TSZ) return;
    int b = p / TSZ, r = p % TSZ;
    size_t base = (size_t)b * EE * TSZ + r;
    float sum = 0.f, sq = 0.f;
    for (int e = 0; e < EE; ++e) {
        float x = xin[base + (size_t)e * TSZ];
        sum += x; sq += x * x;
    }
    float mu = sum * (1.f / EE);
    float var = sq * (1.f / EE) - mu * mu;
    float rs = rsqrtf(var + 1e-5f);
    for (int e = 0; e < EE; ++e) {
        size_t idx = base + (size_t)e * TSZ;
        s2[idx] = f2bf((xin[idx] - mu) * rs * g[e] + bt[e]);
    }
}

// ---------------------------------------------------------------------------
// bf16 WMMA GEMM:  Y[o,p] = act( sum_i W[o,i] * X[i,p] + bias[o] ) (+ residual)
// block tile 64(M) x 128(N), 8 waves of 32x32, K-step 32.
// Double-buffered LDS; tiles staged with global_load_async_to_lds_b128 and the
// next tile's copy overlapped with the current tile's WMMAs (ASYNCcnt sync).
// EPI: 0 = bf16 store, 1 = ReLU + bf16 store, 2 = f32 store with residual add
// ---------------------------------------------------------------------------
template <int EPI>
__global__ __launch_bounds__(256) void gemm_bf16_kernel(
    const unsigned short* __restrict__ W,   // [O x I] bf16 row-major
    const unsigned short* __restrict__ X,   // [B][I x P] bf16
    const float* __restrict__ bias,         // [O]
    const float* __restrict__ R,            // [B][O x P] f32 residual (EPI==2)
    unsigned short* __restrict__ Ybf,
    float* __restrict__ Yf,
    int O, int I, int P) {
    __shared__ unsigned short As[2][64 * 32];    // A tile [m][k]
    __shared__ unsigned short Xb[2][32 * 128];   // X tile [k][p]

    int tid = threadIdx.x;
    size_t xoff = (size_t)blockIdx.z * I * P;
    size_t yoff = (size_t)blockIdx.z * O * P;
    int o0 = blockIdx.y * 64, p0 = blockIdx.x * 128;
    int wave = tid >> 5, lane = tid & 31, hi = lane >> 4, nl = lane & 15;
    int wm = wave >> 2, wn = wave & 3;

    v8f zero = {0.f, 0.f, 0.f, 0.f, 0.f, 0.f, 0.f, 0.f};
    v8f acc[2][2];
#pragma unroll
    for (int i = 0; i < 2; ++i)
#pragma unroll
        for (int j = 0; j < 2; ++j) acc[i][j] = zero;

    // -------- tile staging (async 16B chunks, or coalesced dword fallback) --
    auto stage = [&](int buf, int k0) {
#if USE_ASYNC
        {   // A: 64x32 bf16 = 256 x 16B chunks, one per thread
            int m = tid >> 2, c = (tid & 3) * 8;
            async_cp16(W + (size_t)(o0 + m) * I + k0 + c, &As[buf][m * 32 + c]);
        }
#pragma unroll
        for (int j = 0; j < 2; ++j) {  // X: 32x128 bf16 = 512 x 16B chunks
            int c = tid + j * 256;
            int kk = c >> 4, poff = (c & 15) * 8;
            async_cp16(&X[xoff + (size_t)(k0 + kk) * P + p0 + poff],
                       &Xb[buf][kk * 128 + poff]);
        }
#else
#pragma unroll
        for (int j = 0; j < 4; ++j) {  // A: 1024 dwords
            int idx = tid + j * 256;
            int m = idx >> 4, kp = idx & 15;
            *(unsigned int*)&As[buf][m * 32 + kp * 2] =
                *(const unsigned int*)(W + (size_t)(o0 + m) * I + k0 + kp * 2);
        }
#pragma unroll
        for (int j = 0; j < 8; ++j) {  // X: 2048 dwords
            int idx = tid + j * 256;
            int kk = idx >> 6, pp = (idx & 63) * 2;
            *(unsigned int*)&Xb[buf][kk * 128 + pp] =
                *(const unsigned int*)&X[xoff + (size_t)(k0 + kk) * P + p0 + pp];
        }
#endif
    };

    int nk = I / 32;
    stage(0, 0);
    for (int it = 0; it < nk; ++it) {
        int cur = it & 1;
        async_wait0();
        __syncthreads();
        if (it + 1 < nk) stage(cur ^ 1, (it + 1) * 32);

        Frag a[2], bfr[2];
#pragma unroll
        for (int i = 0; i < 2; ++i) {
            int ml = wm * 32 + i * 16 + nl;
#pragma unroll
            for (int v = 0; v < 8; ++v) {
                int k = (v < 4 ? 2 * v : 16 + 2 * (v - 4)) + hi * 8;
                a[i].u[v] = *(const unsigned int*)&As[cur][ml * 32 + k];
            }
        }
#pragma unroll
        for (int j = 0; j < 2; ++j) {
            int nloc = wn * 32 + j * 16 + nl;
#pragma unroll
            for (int v = 0; v < 8; ++v) {
                int k = (v < 4 ? 2 * v : 16 + 2 * (v - 4)) + hi * 8;
                unsigned int lo = Xb[cur][k * 128 + nloc];
                unsigned int hv = Xb[cur][(k + 1) * 128 + nloc];
                bfr[j].u[v] = lo | (hv << 16);
            }
        }
#pragma unroll
        for (int i = 0; i < 2; ++i)
#pragma unroll
            for (int j = 0; j < 2; ++j)
                acc[i][j] = __builtin_amdgcn_wmma_f32_16x16x32_bf16(
                    false, a[i].v, false, bfr[j].v, (short)0, acc[i][j], false, false);
        __syncthreads();
    }

#pragma unroll
    for (int i = 0; i < 2; ++i)
#pragma unroll
        for (int j = 0; j < 2; ++j)
#pragma unroll
            for (int r = 0; r < 8; ++r) {
                int row = o0 + wm * 32 + i * 16 + r + 8 * hi;
                int col = p0 + wn * 32 + j * 16 + nl;
                float val = acc[i][j][r] + bias[row];
                size_t oidx = yoff + (size_t)row * P + col;
                if (EPI == 0)      Ybf[oidx] = f2bf(val);
                else if (EPI == 1) Ybf[oidx] = f2bf(fmaxf(val, 0.f));
                else               Yf[oidx]  = val + R[oidx];
            }
}

// ---------------------------------------------------------------------------
// Flash attention per (b,h,t): S=512, DH=32, multiplicative distance mask.
// Block = 4 waves; each wave owns 128 query rows (8 tiles of 16).
// K tile staged to LDS via async copies; online softmax; QK^T and P*V through
// v_wmma_f32_16x16x32_bf16.
// ---------------------------------------------------------------------------
__global__ __launch_bounds__(128) void attn_kernel(
    const unsigned short* __restrict__ Q, const unsigned short* __restrict__ K,
    const unsigned short* __restrict__ V, const float* __restrict__ coord,
    unsigned short* __restrict__ vals) {
    __shared__ unsigned short Kl[DHH * SS];        // K staged [d][u], 32 KB
    __shared__ float cxs[SS], cys[SS];             // coords, 4 KB
    __shared__ unsigned short stage[4][16 * 34];   // per-wave P transpose, 4.25 KB

    int tid = threadIdx.x;
    int blk = blockIdx.x;
    int b = blk >> 7;            // / (H*T)
    int h = (blk >> 4) & 7;
    int t = blk & 15;
    size_t base = (size_t)b * EE * TSZ + (size_t)(h * DHH) * TSZ + (size_t)t * SS;
    const unsigned short* Qb = Q + base;
    const unsigned short* Kb = K + base;
    const unsigned short* Vb = V + base;
    unsigned short* Ob = vals + base;
    const float* cb = coord + (size_t)(b * TT + t) * SS * 2;

#if USE_ASYNC
    for (int i = 0; i < 16; ++i) {               // 2048 x 16B chunks / 128 thr
        int c = tid + i * 128;
        int d = c >> 6, uoff = (c & 63) * 8;
        async_cp16(Kb + (size_t)d * TSZ + uoff, &Kl[d * SS + uoff]);
    }
#else
    for (int i = 0; i < 64; ++i) {               // 8192 dwords / 128 threads
        int idx = tid + i * 128;
        int d = idx >> 8, up = (idx & 255) * 2;
        *(unsigned int*)&Kl[d * SS + up] =
            *(const unsigned int*)&Kb[(size_t)d * TSZ + up];
    }
#endif
    for (int i = 0; i < 4; ++i) {
        int s = tid + i * 128;
        cxs[s] = cb[s * 2];
        cys[s] = cb[s * 2 + 1];
    }
    async_wait0();
    __syncthreads();

    int wave = tid >> 5, lane = tid & 31, hi = lane >> 4, nl = lane & 15;
    unsigned short* stg = stage[wave];
    const float scale = 0.17677669529663687f;    // 1/sqrt(32)
    v8f zero = {0.f, 0.f, 0.f, 0.f, 0.f, 0.f, 0.f, 0.f};

    for (int rt = 0; rt < 8; ++rt) {
        int s0 = wave * 128 + rt * 16;
        // Q fragment (16x32, K = full head dim)
        Frag aq;
#pragma unroll
        for (int v = 0; v < 8; ++v) {
            int k = (v < 4 ? 2 * v : 16 + 2 * (v - 4)) + hi * 8;
            int s = s0 + nl;
            unsigned int lo = Qb[(size_t)k * TSZ + s];
            unsigned int hv = Qb[(size_t)(k + 1) * TSZ + s];
            aq.u[v] = lo | (hv << 16);
        }
        float sx[8], sy[8], mrow[8], ssum[8];
#pragma unroll
        for (int r = 0; r < 8; ++r) {
            int s = s0 + r + 8 * hi;
            sx[r] = cxs[s]; sy[r] = cys[s];
            mrow[r] = -1e30f; ssum[r] = 0.f;
        }
        v8f o0 = zero, o1 = zero;

        for (int pr = 0; pr < 16; ++pr) {        // pairs of 16-wide key tiles
            v8f cc[2];
#pragma unroll
            for (int q = 0; q < 2; ++q) {
                int ut = pr * 2 + q;
                Frag bk;
#pragma unroll
                for (int v = 0; v < 8; ++v) {
                    int kk = (v < 4 ? 2 * v : 16 + 2 * (v - 4)) + hi * 8;
                    int u = ut * 16 + nl;
                    unsigned int lo = Kl[kk * SS + u];
                    unsigned int hv = Kl[(kk + 1) * SS + u];
                    bk.u[v] = lo | (hv << 16);
                }
                v8f c = __builtin_amdgcn_wmma_f32_16x16x32_bf16(
                    false, aq.v, false, bk.v, (short)0, zero, false, false);
                int u = ut * 16 + nl;
                float ux = cxs[u], uy = cys[u];
#pragma unroll
                for (int r = 0; r < 8; ++r) {
                    float dx = sx[r] - ux, dy = sy[r] - uy;
                    float vv = c[r] * scale;
                    c[r] = (dx * dx + dy * dy < 625.0f) ? vv : 0.f;  // logits *= mask
                }
                cc[q] = c;
            }
            // online softmax stats (16-lane row reductions)
            float alpha[8];
#pragma unroll
            for (int r = 0; r < 8; ++r) {
                float vm = fmaxf(cc[0][r], cc[1][r]);
                vm = fmaxf(vm, __shfl_xor(vm, 1, 32));
                vm = fmaxf(vm, __shfl_xor(vm, 2, 32));
                vm = fmaxf(vm, __shfl_xor(vm, 4, 32));
                vm = fmaxf(vm, __shfl_xor(vm, 8, 32));
                float mn = fmaxf(mrow[r], vm);
                alpha[r] = __expf(mrow[r] - mn);
                mrow[r] = mn;
                float p0 = __expf(cc[0][r] - mn);
                float p1 = __expf(cc[1][r] - mn);
                cc[0][r] = p0; cc[1][r] = p1;
                float rs = p0 + p1;
                rs += __shfl_xor(rs, 1, 32);
                rs += __shfl_xor(rs, 2, 32);
                rs += __shfl_xor(rs, 4, 32);
                rs += __shfl_xor(rs, 8, 32);
                ssum[r] = ssum[r] * alpha[r] + rs;
            }
            // C-layout -> A-layout via per-wave LDS staging (bf16)
#pragma unroll
            for (int r = 0; r < 8; ++r) {
                int m = r + 8 * hi;
                stg[m * 34 + nl]      = f2bf(cc[0][r]);
                stg[m * 34 + 16 + nl] = f2bf(cc[1][r]);
            }
            Frag ap;
#pragma unroll
            for (int v = 0; v < 8; ++v) {
                int k = (v < 4 ? 2 * v : 16 + 2 * (v - 4)) + hi * 8;
                ap.u[v] = *(const unsigned int*)&stg[nl * 34 + k];
            }
#pragma unroll
            for (int r = 0; r < 8; ++r) { o0[r] *= alpha[r]; o1[r] *= alpha[r]; }
            // V fragments: k-pairs (u,u+1) are contiguous dwords in [e][p] layout
            Frag bv0, bv1;
#pragma unroll
            for (int v = 0; v < 8; ++v) {
                int kk = (v < 4 ? 2 * v : 16 + 2 * (v - 4)) + hi * 8;
                int u = pr * 32 + kk;
                bv0.u[v] = *(const unsigned int*)&Vb[(size_t)nl * TSZ + u];
                bv1.u[v] = *(const unsigned int*)&Vb[(size_t)(16 + nl) * TSZ + u];
            }
            o0 = __builtin_amdgcn_wmma_f32_16x16x32_bf16(false, ap.v, false, bv0.v,
                                                         (short)0, o0, false, false);
            o1 = __builtin_amdgcn_wmma_f32_16x16x32_bf16(false, ap.v, false, bv1.v,
                                                         (short)0, o1, false, false);
        }
#pragma unroll
        for (int r = 0; r < 8; ++r) {
            float inv = 1.f / ssum[r];
            int s = s0 + r + 8 * hi;
            Ob[(size_t)nl * TSZ + s]        = f2bf(o0[r] * inv);
            Ob[(size_t)(16 + nl) * TSZ + s] = f2bf(o1[r] * inv);
        }
    }
}

// ---------------------------------------------------------------------------
extern "C" void kernel_launch(void* const* d_in, const int* in_sizes, int n_in,
                              void* d_out, int out_size, void* d_ws, size_t ws_size,
                              hipStream_t stream) {
    (void)in_sizes; (void)n_in; (void)out_size; (void)ws_size;
    const float* enc  = (const float*)d_in[0];
    const float* pos  = (const float*)d_in[1];
    const float* crd  = (const float*)d_in[2];
    const float* ln1g = (const float*)d_in[3];
    const float* ln1b = (const float*)d_in[4];
    const float* ln2g = (const float*)d_in[5];
    const float* ln2b = (const float*)d_in[6];
    const float* Wq = (const float*)d_in[7];  const float* bq = (const float*)d_in[8];
    const float* Wk = (const float*)d_in[9];  const float* bk = (const float*)d_in[10];
    const float* Wv = (const float*)d_in[11]; const float* bv = (const float*)d_in[12];
    const float* Wo = (const float*)d_in[13]; const float* bo = (const float*)d_in[14];
    const float* W1 = (const float*)d_in[15]; const float* b1 = (const float*)d_in[16];
    const float* W2 = (const float*)d_in[17]; const float* b2 = (const float*)d_in[18];
    float* out = (float*)d_out;

    char* ws = (char*)d_ws;
    size_t off = 0;
    auto alloc = [&](size_t bytes) -> void* {
        void* p = ws + off;
        off += (bytes + 255) & ~(size_t)255;
        return p;
    };
    const size_t NTOT = (size_t)BB * EE * TSZ;                 // 4.19M elems
    unsigned short* wq_bf = (unsigned short*)alloc((size_t)EE * EE * 2);
    unsigned short* wk_bf = (unsigned short*)alloc((size_t)EE * EE * 2);
    unsigned short* wv_bf = (unsigned short*)alloc((size_t)EE * EE * 2);
    unsigned short* wo_bf = (unsigned short*)alloc((size_t)EE * EE * 2);
    unsigned short* w1_bf = (unsigned short*)alloc((size_t)FF * EE * 2);
    unsigned short* w2_bf = (unsigned short*)alloc((size_t)EE * FF * 2);
    unsigned short* qkbuf = (unsigned short*)alloc(NTOT * 2);  // later reused for vals
    unsigned short* s2buf = (unsigned short*)alloc(NTOT * 2);
    unsigned short* Qb    = (unsigned short*)alloc(NTOT * 2);
    unsigned short* Kb    = (unsigned short*)alloc(NTOT * 2);
    unsigned short* Vb    = (unsigned short*)alloc(NTOT * 2);
    unsigned short* ffh   = (unsigned short*)alloc((size_t)BB * FF * TSZ * 2);
    float*          xws   = (float*)alloc(NTOT * 4);

    // 1) weights -> bf16
    {
        int n = EE * EE;
        cvt_bf16_kernel<<<(n + 255) / 256, 256, 0, stream>>>(Wq, wq_bf, n);
        cvt_bf16_kernel<<<(n + 255) / 256, 256, 0, stream>>>(Wk, wk_bf, n);
        cvt_bf16_kernel<<<(n + 255) / 256, 256, 0, stream>>>(Wv, wv_bf, n);
        cvt_bf16_kernel<<<(n + 255) / 256, 256, 0, stream>>>(Wo, wo_bf, n);
        int nf = FF * EE;
        cvt_bf16_kernel<<<(nf + 255) / 256, 256, 0, stream>>>(W1, w1_bf, nf);
        cvt_bf16_kernel<<<(nf + 255) / 256, 256, 0, stream>>>(W2, w2_bf, nf);
    }
    // 2) LN1 (+pos)
    ln1_kernel<<<(BB * TSZ) / 256, 256, 0, stream>>>(enc, pos, ln1g, ln1b, qkbuf, s2buf);
    // 3-5) q/k/v projections
    dim3 gE(TSZ / 128, EE / 64, BB);
    gemm_bf16_kernel<0><<<gE, 256, 0, stream>>>(wq_bf, qkbuf, bq, nullptr, Qb, nullptr, EE, EE, TSZ);
    gemm_bf16_kernel<0><<<gE, 256, 0, stream>>>(wk_bf, qkbuf, bk, nullptr, Kb, nullptr, EE, EE, TSZ);
    gemm_bf16_kernel<0><<<gE, 256, 0, stream>>>(wv_bf, s2buf, bv, nullptr, Vb, nullptr, EE, EE, TSZ);
    // 6) attention -> vals (reuses qkbuf)
    attn_kernel<<<BB * HH * TT, 128, 0, stream>>>(Qb, Kb, Vb, crd, qkbuf);
    // 7) output projection + residual -> x (f32)
    gemm_bf16_kernel<2><<<gE, 256, 0, stream>>>(wo_bf, qkbuf, bo, enc, nullptr, xws, EE, EE, TSZ);
    // 8) LN2
    ln2_kernel<<<(BB * TSZ) / 256, 256, 0, stream>>>(xws, ln2g, ln2b, s2buf);
    // 9) FFN up + ReLU
    dim3 gF(TSZ / 128, FF / 64, BB);
    gemm_bf16_kernel<1><<<gF, 256, 0, stream>>>(w1_bf, s2buf, b1, nullptr, ffh, nullptr, FF, EE, TSZ);
    // 10) FFN down + residual -> out
    gemm_bf16_kernel<2><<<gE, 256, 0, stream>>>(w2_bf, ffh, b2, xws, nullptr, out, EE, FF, TSZ);
}
